// GRUT1_84593675862254
// MI455X (gfx1250) — matrix operand order repaired and checked by reading, not compiled, not run on hardware
//
#include <hip/hip_runtime.h>

// Time-aware GRU scan for MI455X (gfx1250, wave32, WMMA).
// One wave owns a 16-row batch tile, iterates T=512 steps.
// Recurrent + input projections fused into bf16 WMMA (f32 accumulate).
// All 36 weight B-fragments pinned in VGPRs for the whole scan
// (gfx1250 extended VGPR file, occupancy-1 by design).
// Hidden-unit permutation pi(16j+l) = l*4+j makes the per-step
// C-layout -> A-layout staging a packed ds_store_b64 scatter.

#define F_IN  5
#define HID   64
#define TLEN  512
#define BSZ   4096
#define NB_TILE 16
#define WAVES_PER_BLOCK 4

typedef __attribute__((ext_vector_type(16))) __bf16       bf16x16;
typedef __attribute__((ext_vector_type(8)))  float        f32x8;
typedef __attribute__((ext_vector_type(4)))  unsigned int u32x4;
typedef __attribute__((ext_vector_type(2)))  unsigned int u32x2;

union Frag { u32x4 u[2]; bf16x16 v; };

__device__ __forceinline__ unsigned int f2bf(float f) {
  __bf16 b = (__bf16)f;
  return (unsigned int)__builtin_bit_cast(unsigned short, b);
}

__device__ __forceinline__ unsigned int pack_bf16(float lo, float hi) {
  return (f2bf(hi) << 16) | f2bf(lo);
}

__device__ __forceinline__ float fast_sigmoid(float x) {
  return 1.0f / (1.0f + __expf(-x));
}

__device__ __forceinline__ float fast_tanh(float x) {
  float e = __expf(2.0f * fabsf(x));
  float t = 1.0f - 2.0f / (e + 1.0f);
  return copysignf(t, x);
}

__global__ __launch_bounds__(WAVES_PER_BLOCK * 32, 1)
void GRUT1_kernel(const float* __restrict__ X,   const float* __restrict__ dt,
                  const float* __restrict__ w_ih, const float* __restrict__ w_hh,
                  const float* __restrict__ b_ih, const float* __restrict__ b_hh,
                  const float* __restrict__ w_dt, const float* __restrict__ b_dt,
                  const float* __restrict__ w_out, const float* __restrict__ b_out,
                  float* __restrict__ out)
{
  // Per-wave h_tilde staging tile (row-major 16x64 bf16 in permuted-K space).
  __shared__ unsigned short sH[WAVES_PER_BLOCK][NB_TILE * HID]; // 8 KB

  const int tid   = threadIdx.x;
  const int wave  = tid >> 5;
  const int lane  = tid & 31;
  const int lhalf = lane >> 4;   // 0: lanes 0-15, 1: lanes 16-31
  const int lmod  = lane & 15;
  const int b0    = (blockIdx.x * WAVES_PER_BLOCK + wave) * NB_TILE;

  unsigned short* hs = &sH[wave][0];   // [m][k'] bf16, 16x64

  const u32x4 z4   = {0u, 0u, 0u, 0u};
  const f32x8 zacc = {0.f, 0.f, 0.f, 0.f, 0.f, 0.f, 0.f, 0.f};

  // ---- Build all B fragments directly from global into registers ----
  // B-layout (32x16 bf16): lane = lhalf*16 + n ; value i -> k' = lhalf*16 + i.
  // Hidden K is permuted: k' = pi(col), pi(16j + l) = l*4 + j, so
  // pi^-1(k') = 16*(k' & 3) + (k' >> 2).
  //   Bh0[g][j]: w_hh[row, pi^-1(lhalf*16 + i)]        (k' = 0..31)
  //   Bh1[g][j]: w_hh[row, pi^-1(32 + lhalf*16 + i)]   (k' = 32..63)
  //   Bx [g][j]: w_ih[row, 0..4] zero-padded to K=32   (x chunk, unpermuted)
  // with row = g*64 + j*16 + n, n = lmod.
  bf16x16 Bx[3][4], Bh0[3][4], Bh1[3][4];
#pragma unroll
  for (int g = 0; g < 3; ++g) {
#pragma unroll
    for (int j = 0; j < 4; ++j) {
      const int row = g * 64 + j * 16 + lmod;
      const float* wr = w_hh + row * HID;
      Frag f0, f1;
#pragma unroll
      for (int i = 0; i < 16; ++i) {
        const int k0 = lhalf * 16 + i;        // k' in chunk 0
        const int k1 = 32 + lhalf * 16 + i;   // k' in chunk 1
        f0.v[i] = (__bf16)wr[16 * (k0 & 3) + (k0 >> 2)];
        f1.v[i] = (__bf16)wr[16 * (k1 & 3) + (k1 >> 2)];
      }
      Bh0[g][j] = f0.v;
      Bh1[g][j] = f1.v;

      Frag fx; fx.u[0] = z4; fx.u[1] = z4;
      if (lhalf == 0) {                  // only K=0..7 lanes carry x weights
        const float* xr = w_ih + row * F_IN;
#pragma unroll
        for (int i = 0; i < F_IN; ++i) fx.v[i] = (__bf16)xr[i];
      }
      Bx[g][j] = fx.v;
    }
  }

  // Per-lane column constants; in C-layout this lane owns column (16j + lmod).
  float wdt[4], bdt[4], br[4], bz[4], bin[4], bhn[4], wout[4];
#pragma unroll
  for (int j = 0; j < 4; ++j) {
    int col = 16 * j + lmod;
    wdt[j]  = w_dt[col];
    bdt[j]  = b_dt[col];
    br[j]   = b_ih[col]       + b_hh[col];        // r gate bias (f32, exact)
    bz[j]   = b_ih[64 + col]  + b_hh[64 + col];   // z gate bias
    bin[j]  = b_ih[128 + col];                    // n gate input bias
    bhn[j]  = b_hh[128 + col];                    // n gate hidden bias (inside r*)
    wout[j] = w_out[col];
  }
  const float bo = b_out[0];

  // h in f32 C-layout: h[j][v] = h[M = v + 8*lhalf, col = 16j + lmod]
  float h[4][8];
#pragma unroll
  for (int j = 0; j < 4; ++j)
#pragma unroll
    for (int v = 0; v < 8; ++v) h[j][v] = 0.0f;

#pragma clang loop unroll(disable)
  for (int t = 0; t < TLEN; ++t) {
    // ---- x A-fragment (16x32 bf16): A[m, k<5] = X[b0+m, t, k], rest 0 ----
    Frag axf; axf.u[0] = z4; axf.u[1] = z4;
    if (lhalf == 0) {   // lanes 0-15 hold K=0..7 -> only they carry x
      const float* xp = X + ((size_t)(b0 + lmod) * TLEN + t) * F_IN;
#pragma unroll
      for (int f = 0; f < F_IN; ++f) axf.v[f] = (__bf16)xp[f];
    }

    // ---- dt for the 8 batch rows (M) this lane covers in C-layout ----
    float dtv[8];
#pragma unroll
    for (int v = 0; v < 8; ++v)
      dtv[v] = dt[(size_t)(b0 + v + 8 * lhalf) * TLEN + t];

    // ---- decay gate, h_tilde; stage to LDS in permuted-K space ----
    // Lane's 4 j-values for row m land at k' = lmod*4 + j (contiguous):
    // one packed ds_store_b64 per v.
    float ht[4][8];
#pragma unroll
    for (int v = 0; v < 8; ++v) {
      float e[4];
#pragma unroll
      for (int j = 0; j < 4; ++j) {
        float decay = fmaxf(dtv[v] * wdt[j] + bdt[j], 0.0f);
        float hv = __expf(-decay) * h[j][v];
        ht[j][v] = hv;
        e[j] = hv;
      }
      u32x2 pk;
      pk.x = pack_bf16(e[0], e[1]);
      pk.y = pack_bf16(e[2], e[3]);
      *(u32x2*)(hs + (v + 8 * lhalf) * HID + lmod * 4) = pk;
    }
    asm volatile("s_wait_dscnt 0x0" ::: "memory");  // LDS scatter -> gather fence

    // ---- gather h_tilde A-fragments (16x32 each, permuted-K space) ----
    // chunk0: k' = lhalf*16 + 0..15 ; chunk1: +32
    const unsigned short* hrow = hs + lmod * HID + lhalf * 16;
    Frag a0, a1;
    a0.u[0] = *(const u32x4*)(hrow);
    a0.u[1] = *(const u32x4*)(hrow + 8);
    a1.u[0] = *(const u32x4*)(hrow + 32);
    a1.u[1] = *(const u32x4*)(hrow + 40);

#pragma unroll
    for (int j = 0; j < 4; ++j) {
      // r gate: x-chunk + two h-chunks chained into one f32 accumulator
      f32x8 accr = __builtin_amdgcn_wmma_f32_16x16x32_bf16(
          false, a1.v, false, Bh1[0][j], (short)0, zacc, false, false);
      accr = __builtin_amdgcn_wmma_f32_16x16x32_bf16(
          false, a0.v, false, Bh0[0][j], (short)0, accr, false, false);
      accr = __builtin_amdgcn_wmma_f32_16x16x32_bf16(
          false, axf.v, false, Bx[0][j], (short)0, accr, false, false);

      // z gate
      f32x8 accz = __builtin_amdgcn_wmma_f32_16x16x32_bf16(
          false, a1.v, false, Bh1[1][j], (short)0, zacc, false, false);
      accz = __builtin_amdgcn_wmma_f32_16x16x32_bf16(
          false, a0.v, false, Bh0[1][j], (short)0, accz, false, false);
      accz = __builtin_amdgcn_wmma_f32_16x16x32_bf16(
          false, axf.v, false, Bx[1][j], (short)0, accz, false, false);

      // n gate: hidden part and input part kept separate (r multiplies hidden part)
      f32x8 accnh = __builtin_amdgcn_wmma_f32_16x16x32_bf16(
          false, a1.v, false, Bh1[2][j], (short)0, zacc, false, false);
      accnh = __builtin_amdgcn_wmma_f32_16x16x32_bf16(
          false, a0.v, false, Bh0[2][j], (short)0, accnh, false, false);
      f32x8 accni = __builtin_amdgcn_wmma_f32_16x16x32_bf16(
          false, axf.v, false, Bx[2][j], (short)0, zacc, false, false);

      // elementwise gate math in f32 (C-layout)
#pragma unroll
      for (int v = 0; v < 8; ++v) {
        float rr = fast_sigmoid(accr[v] + br[j]);
        float zz = fast_sigmoid(accz[v] + bz[j]);
        float nn = fast_tanh(accni[v] + bin[j] + rr * (accnh[v] + bhn[j]));
        h[j][v] = (1.0f - zz) * nn + zz * ht[j][v];
      }
    }

    // ---- output projection: pred[b,t] = sum_col h * w_out[col] + b_out ----
    float p[8];
#pragma unroll
    for (int v = 0; v < 8; ++v)
      p[v] = h[0][v] * wout[0] + h[1][v] * wout[1] +
             h[2][v] * wout[2] + h[3][v] * wout[3];
    // 16-lane butterfly (masks < 16 stay within each C-layout half)
#pragma unroll
    for (int m = 1; m < 16; m <<= 1) {
#pragma unroll
      for (int v = 0; v < 8; ++v) p[v] += __shfl_xor(p[v], m, 32);
    }
#pragma unroll
    for (int v = 0; v < 8; ++v) {
      if (lmod == v)
        out[(size_t)(b0 + v + 8 * lhalf) * TLEN + t] = p[v] + bo;
    }
  }
}

extern "C" void kernel_launch(void* const* d_in, const int* in_sizes, int n_in,
                              void* d_out, int out_size, void* d_ws, size_t ws_size,
                              hipStream_t stream) {
  (void)in_sizes; (void)n_in; (void)out_size; (void)d_ws; (void)ws_size;
  const float* X     = (const float*)d_in[0];
  const float* dtp   = (const float*)d_in[1];
  const float* w_ih  = (const float*)d_in[2];
  const float* w_hh  = (const float*)d_in[3];
  const float* b_ih  = (const float*)d_in[4];
  const float* b_hh  = (const float*)d_in[5];
  const float* w_dt  = (const float*)d_in[6];
  const float* b_dt  = (const float*)d_in[7];
  const float* w_out = (const float*)d_in[8];
  const float* b_out = (const float*)d_in[9];
  float* out = (float*)d_out;

  dim3 grid(BSZ / (WAVES_PER_BLOCK * NB_TILE));   // 64 workgroups
  dim3 block(WAVES_PER_BLOCK * 32);               // 128 threads = 4 waves
  hipLaunchKernelGGL(GRUT1_kernel, grid, block, 0, stream,
                     X, dtp, w_ih, w_hh, b_ih, b_hh, w_dt, b_dt, w_out, b_out, out);
}
